// GNNEncoder_21887153340886
// MI455X (gfx1250) — matrix operand order repaired and checked by reading, not compiled, not run on hardware
//
#include <hip/hip_runtime.h>

#define N_NODES 40000
#define N_EDGES 640000
#define DIM 128

typedef __attribute__((ext_vector_type(16))) _Float16 v16h;
typedef __attribute__((ext_vector_type(8)))  _Float16 v8h;
typedef __attribute__((ext_vector_type(8)))  float    v8f;

// ---------------- workspace layout (bytes) ----------------
// mean   : [0,           20480000)   40000*128 f32 neighbor-mean
// deg    : [20480000,    +160000)    40000 u32 in-degree
// cursor : [20640000,    +160000)    40000 u32 fill cursor
// rowst  : [20800000,    +160256)    40001 u32 CSR row starts
// neigh  : [20960256,    +2560000)   640000 u32 CSR column (src) indices
// Wp1    : [23520256,    +65536)     layer1 packed f16 weights (B-frag layout)
// Wp2    : [23585792,    +65536)     layer2 packed f16 weights
#define OFF_DEG  20480000u
#define OFF_CUR  20640000u
#define OFF_RS   20800000u
#define OFF_NB   20960256u
#define OFF_WP1  23520256u
#define OFF_WP2  23585792u

// -------- zero a u32 range --------
__global__ void zero_u32_k(unsigned* __restrict__ p, int n) {
    int t = blockIdx.x * 256 + threadIdx.x;
    if (t < n) p[t] = 0u;
}

// -------- per-edge in-degree count (integer atomics, deterministic) --------
__global__ void count_k(const long long* __restrict__ ei, unsigned* __restrict__ deg) {
    int e = blockIdx.x * 256 + threadIdx.x;
    if (e < N_EDGES) {
        int dst = (int)ei[N_EDGES + e];
        atomicAdd(&deg[dst], 1u);
    }
}

// -------- exclusive prefix sum of deg -> rowst (single 1024-thread block) --------
__global__ __launch_bounds__(1024) void scan_k(const unsigned* __restrict__ deg,
                                               unsigned* __restrict__ rs) {
    __shared__ unsigned s[1024];
    __shared__ unsigned carry;
    const int t = threadIdx.x;
    if (t == 0) carry = 0u;
    __syncthreads();
    for (int base = 0; base < N_NODES; base += 1024) {
        int i = base + t;
        unsigned v = (i < N_NODES) ? deg[i] : 0u;
        s[t] = v;
        __syncthreads();
        for (int off = 1; off < 1024; off <<= 1) {
            unsigned add = (t >= off) ? s[t - off] : 0u;
            __syncthreads();
            s[t] += add;
            __syncthreads();
        }
        unsigned incl = s[t];
        unsigned run = carry;           // stable: last write was before a barrier
        if (i < N_NODES) rs[i] = run + incl - v;
        __syncthreads();
        if (t == 1023) carry = run + incl;
        __syncthreads();
    }
    if (t == 0) rs[N_NODES] = carry;    // == N_EDGES
}

// -------- CSR bucket fill --------
__global__ void fill_k(const long long* __restrict__ ei,
                       const unsigned* __restrict__ rs,
                       unsigned* __restrict__ cur,
                       unsigned* __restrict__ nb) {
    int e = blockIdx.x * 256 + threadIdx.x;
    if (e < N_EDGES) {
        int src = (int)ei[e];
        int dst = (int)ei[N_EDGES + e];
        unsigned pos = atomicAdd(&cur[dst], 1u);
        nb[rs[dst] + pos] = (unsigned)src;
    }
}

// -------- per-node neighbor-list sort (fixes FP reduction order -> bit-deterministic) --------
__global__ void sort_k(const unsigned* __restrict__ rs, unsigned* __restrict__ nb) {
    int i = blockIdx.x * 256 + threadIdx.x;
    if (i < N_NODES) {
        unsigned a = rs[i], b = rs[i + 1];
        for (unsigned j = a + 1; j < b; ++j) {
            unsigned key = nb[j];
            unsigned k = j;
            while (k > a && nb[k - 1] > key) { nb[k] = nb[k - 1]; --k; }
            nb[k] = key;
        }
    }
}

// -------- pack [Wl;Wr] (each 128x128 f32, row=out-feature) into f16 B-fragment layout --------
// flat f16 index t = ((nt*8 + ks)*32 + lane)*16 + i
//   n = nt*16 + (lane&15), k = ks*32 + (lane>>4)*16 + i   (k in [0,256))
// B[k][n] = k<128 ? Wl[n][k] : Wr[n][k-128]
__global__ void prep_w(const float* __restrict__ Wl, const float* __restrict__ Wr,
                       _Float16* __restrict__ Wp) {
    int t = blockIdx.x * 256 + threadIdx.x;      // 32768 total
    int i    =  t        & 15;
    int lane = (t >> 4)  & 31;
    int ks   = (t >> 9)  & 7;
    int nt   =  t >> 12;                          // 0..7
    int n = nt * 16 + (lane & 15);
    int k = ks * 32 + (lane >> 4) * 16 + i;
    float v = (k < DIM) ? Wl[n * DIM + k] : Wr[n * DIM + (k - DIM)];
    Wp[t] = (_Float16)v;
}

// -------- gather mean-aggregation: one wave per dst node, float4 per lane --------
__global__ __launch_bounds__(256) void agg_mean_k(const unsigned* __restrict__ rs,
                                                  const unsigned* __restrict__ nb,
                                                  const float* __restrict__ feat,
                                                  float* __restrict__ mean) {
    int t = blockIdx.x * 256 + threadIdx.x;       // 5000 blocks exactly
    int node = t >> 5;
    int lane = t & 31;
    unsigned a = rs[node], b = rs[node + 1];
    float4 acc0 = make_float4(0.f, 0.f, 0.f, 0.f);
    float4 acc1 = make_float4(0.f, 0.f, 0.f, 0.f);
    unsigned j = a;
    for (; j + 2 <= b; j += 2) {
        unsigned s0 = nb[j], s1 = nb[j + 1];
        float4 v0 = ((const float4*)(feat + (size_t)s0 * DIM))[lane];
        float4 v1 = ((const float4*)(feat + (size_t)s1 * DIM))[lane];
        acc0.x += v0.x; acc0.y += v0.y; acc0.z += v0.z; acc0.w += v0.w;
        acc1.x += v1.x; acc1.y += v1.y; acc1.z += v1.z; acc1.w += v1.w;
    }
    if (j < b) {
        unsigned s0 = nb[j];
        float4 v0 = ((const float4*)(feat + (size_t)s0 * DIM))[lane];
        acc0.x += v0.x; acc0.y += v0.y; acc0.z += v0.z; acc0.w += v0.w;
    }
    unsigned d = b - a;
    float sc = 1.0f / (float)(d ? d : 1u);
    float4 r;
    r.x = (acc0.x + acc1.x) * sc;
    r.y = (acc0.y + acc1.y) * sc;
    r.z = (acc0.z + acc1.z) * sc;
    r.w = (acc0.w + acc1.w) * sc;
    ((float4*)(mean + (size_t)node * DIM))[lane] = r;
}

// -------- fused SAGE layer GEMM: out = relu([mean | xin](N,256) @ Bpacked(256,128) + bias) --------
// block = 16 rows, 256 threads = 8 waves; wave w owns output N-tile [w*16, w*16+16)
#define AS_STRIDE 264   // 256 f16 + 8 pad => 528B row stride (breaks LDS bank alignment)
__global__ __launch_bounds__(256) void sage_gemm_k(
        const float*    __restrict__ mean,
        const float*    __restrict__ xin,
        const _Float16* __restrict__ Wp,
        const float*    __restrict__ bias,
        float*          __restrict__ out) {
    __shared__ _Float16 As[16 * AS_STRIDE];

    const int row0 = blockIdx.x * 16;
    const int t    = threadIdx.x;

    // hint: keep packed weights hot (gfx1250 global_prefetch_b8)
    __builtin_prefetch(Wp + (size_t)(t >> 5) * 8 * 32 * 16, 0, 0);

    // ---- stage A = [mean | x] rows row0..row0+15 as f16 into LDS ----
    {
        const int m   = t >> 4;       // row within tile
        const int seg = t & 15;       // 16-col segment
        const int row = row0 + m;
        const float* srcp = ((seg < 8) ? mean : xin) + (size_t)row * DIM;
        const int col = (seg & 7) * 16;
        _Float16* dst = As + m * AS_STRIDE + seg * 16;
#pragma unroll
        for (int j = 0; j < 16; j += 4) {
            float4 v = *(const float4*)(srcp + col + j);
            dst[j + 0] = (_Float16)v.x;
            dst[j + 1] = (_Float16)v.y;
            dst[j + 2] = (_Float16)v.z;
            dst[j + 3] = (_Float16)v.w;
        }
    }
    __syncthreads();

    const int lane = t & 31;
    const int wave = t >> 5;          // = output N-tile index (0..7)
    const int half = lane >> 4;       // 0: lanes 0-15, 1: lanes 16-31
    const int mm   = lane & 15;

    v8f c = {};
    const v16h* Bp = reinterpret_cast<const v16h*>(Wp) + (size_t)wave * 8 * 32;

#pragma unroll
    for (int ks = 0; ks < 8; ++ks) {
        // A fragment (contiguous k order, same as packed B):
        //   elems 0-7  = A[mm][ks*32 + half*16 .. +7]
        //   elems 8-15 = A[mm][ks*32 + half*16 + 8 .. +7]
        const _Float16* ap = As + mm * AS_STRIDE + ks * 32 + half * 16;
        v8h alo = *(const v8h*)(ap);
        v8h ahi = *(const v8h*)(ap + 8);
        v16h a = __builtin_shufflevector(alo, ahi,
                                         0, 1, 2, 3, 4, 5, 6, 7,
                                         8, 9, 10, 11, 12, 13, 14, 15);
        // B fragment: pre-packed, 32B contiguous per lane
        v16h b = Bp[ks * 32 + lane];
        c = __builtin_amdgcn_wmma_f32_16x16x32_f16(
                /*neg_a=*/false, a, /*neg_b=*/false, b,
                /*c_mod=*/(short)0, c, /*reuse_a=*/false, /*reuse_b=*/false);
    }

    // ---- epilogue: bias + relu; c[r] holds D[M = r + half*8][N = mm] ----
    const int n  = wave * 16 + mm;
    const float bn = bias[n];
#pragma unroll
    for (int r = 0; r < 8; ++r) {
        int orow = row0 + r + half * 8;
        float val = c[r] + bn;
        out[(size_t)orow * DIM + n] = val > 0.0f ? val : 0.0f;
    }
}

extern "C" void kernel_launch(void* const* d_in, const int* in_sizes, int n_in,
                              void* d_out, int out_size, void* d_ws, size_t ws_size,
                              hipStream_t stream) {
    const float*     x   = (const float*)d_in[0];
    const long long* ei  = (const long long*)d_in[1];   // int64 edge_index (2, E)
    const float*     W1l = (const float*)d_in[2];
    const float*     b1l = (const float*)d_in[3];
    const float*     W1r = (const float*)d_in[4];
    const float*     W2l = (const float*)d_in[5];
    const float*     b2l = (const float*)d_in[6];
    const float*     W2r = (const float*)d_in[7];
    float* out = (float*)d_out;

    char* ws = (char*)d_ws;
    float*     mean = (float*)ws;
    unsigned*  deg  = (unsigned*)(ws + OFF_DEG);
    unsigned*  cur  = (unsigned*)(ws + OFF_CUR);
    unsigned*  rs   = (unsigned*)(ws + OFF_RS);
    unsigned*  nb   = (unsigned*)(ws + OFF_NB);
    _Float16*  Wp1  = (_Float16*)(ws + OFF_WP1);
    _Float16*  Wp2  = (_Float16*)(ws + OFF_WP2);

    const int EB = (N_EDGES + 255) / 256;   // 2500
    const int NB = (N_NODES + 255) / 256;   // 157

    // ---- build CSR once per call (graph identical for both layers) ----
    zero_u32_k<<<(2 * N_NODES + 255) / 256, 256, 0, stream>>>(deg, 2 * N_NODES); // deg+cur
    count_k   <<<EB, 256, 0, stream>>>(ei, deg);
    scan_k    <<<1, 1024, 0, stream>>>(deg, rs);
    fill_k    <<<EB, 256, 0, stream>>>(ei, rs, cur, nb);
    sort_k    <<<NB, 256, 0, stream>>>(rs, nb);

    // ---- pack weights ----
    prep_w<<<32768 / 256, 256, 0, stream>>>(W1l, W1r, Wp1);
    prep_w<<<32768 / 256, 256, 0, stream>>>(W2l, W2r, Wp2);

    // ---- layer 1: h = relu(mean(x) @ W1l^T + b1l + x @ W1r^T), h stored in d_out ----
    agg_mean_k <<<(N_NODES * 32) / 256, 256, 0, stream>>>(rs, nb, x, mean);
    sage_gemm_k<<<N_NODES / 16, 256, 0, stream>>>(mean, x, Wp1, b1l, out);

    // ---- layer 2: out = relu(mean(h) @ W2l^T + b2l + h @ W2r^T) ----
    agg_mean_k <<<(N_NODES * 32) / 256, 256, 0, stream>>>(rs, nb, out, mean);
    // safe in-place: each block stages its own 16 rows of h into LDS (syncthreads)
    // before overwriting those same rows in d_out; no cross-row reads of h.
    sage_gemm_k<<<N_NODES / 16, 256, 0, stream>>>(mean, out, Wp2, b2l, out);
}
